// GMAN_54468775247906
// MI455X (gfx1250) — matrix operand order repaired and checked by reading, not compiled, not run on hardware
//
#include <hip/hip_runtime.h>

// ---------- types ----------
typedef __bf16 bf16_t;
typedef __attribute__((ext_vector_type(16))) __bf16 v16bf;
typedef __attribute__((ext_vector_type(8)))  float  v8f;
typedef __attribute__((ext_vector_type(4)))  unsigned uint4v;
typedef __attribute__((ext_vector_type(4)))  int     int4v;
typedef __attribute__((ext_vector_type(4)))  float  float4v;
typedef __attribute__((ext_vector_type(4)))  unsigned short ushort4v;

union Frag16 { v16bf v; uint4v q[2]; };

__device__ __forceinline__ unsigned short f2bfbits(float f) {
  unsigned u = __builtin_bit_cast(unsigned, f);
  return (unsigned short)((u + 0x7FFFu + ((u >> 16) & 1u)) >> 16);
}
__device__ __forceinline__ bf16_t f2bf(float f) {
  unsigned short s = f2bfbits(f);
  return __builtin_bit_cast(bf16_t, s);
}
__device__ __forceinline__ v8f wmma_bf16(v16bf a, v16bf b, v8f c) {
  return __builtin_amdgcn_wmma_f32_16x16x32_bf16(false, a, false, b, (short)0, c, false, false);
}

// CDNA5 async global->LDS copy path (guarded: falls back to sync copy if absent)
#if defined(__has_builtin)
#  if __has_builtin(__builtin_amdgcn_global_load_async_to_lds_b128) && \
      __has_builtin(__builtin_amdgcn_s_wait_asynccnt)
#    define ATTN_ASYNC 1
#  endif
#endif
#ifndef ATTN_ASYNC
#  define ATTN_ASYNC 0
#endif

// probed signature: (global int4v* src, shared int4v* dst, int offset, int cpol)
#define AS1_I4(p) ((__attribute__((address_space(1))) int4v*)(p))
#define AS3_I4(p) ((__attribute__((address_space(3))) int4v*)(p))

#define NNODE 325
#define DMODEL 128
#define NROWS 31200   // B*T*N = 8*12*325
#define NQ 336        // 21 * 16 (col tiles)
#define NPADK 352     // 11 * 32 (K pad for P@V)

// ---------- weight pre-pack: fp32 [K,N] -> bf16 B-fragment layout ----------
// P[((ks*N + n)*2 + h)*8 + j] = pack(W[32ks+16h+2j, n], W[32ks+16h+2j+1, n])
__global__ void pack_kernel(const float* __restrict__ W, unsigned* __restrict__ P,
                            int K, int N) {
  int idx = blockIdx.x * 256 + threadIdx.x;
  int total = (K >> 5) * N * 16;
  if (idx >= total) return;
  int j = idx & 7; int h = (idx >> 3) & 1; int rem = idx >> 4;
  int n = rem % N; int ks = rem / N;
  int k0 = ks * 32 + 16 * h + 2 * j;
  unsigned lo = f2bfbits(W[(size_t)k0 * N + n]);
  unsigned hi = f2bfbits(W[(size_t)(k0 + 1) * N + n]);
  P[idx] = lo | (hi << 16);
}

// ---------- bias matrix: alpha*softmax(relu(E E^T)) + beta*lap + mask ----------
__global__ __launch_bounds__(128) void bias_kernel(
    const float* __restrict__ lap, const float* __restrict__ emb,
    const float* __restrict__ alpha_p, const float* __restrict__ beta_p,
    float* __restrict__ bm) {
  const int N = NNODE, E = 16;
  int n = blockIdx.x; int tid = threadIdx.x;
  __shared__ float en[16];
  __shared__ float vals[NNODE];
  __shared__ float red[128];
  if (tid < 16) en[tid] = emb[n * E + tid];
  __syncthreads();
  float lmax = 0.0f;
  for (int m = tid; m < N; m += 128) {
    float d = 0.f;
    #pragma unroll
    for (int j = 0; j < 16; ++j) d += en[j] * emb[m * E + j];
    d = fmaxf(d, 0.f);
    vals[m] = d; lmax = fmaxf(lmax, d);
  }
  red[tid] = lmax; __syncthreads();
  for (int s = 64; s > 0; s >>= 1) { if (tid < s) red[tid] = fmaxf(red[tid], red[tid + s]); __syncthreads(); }
  float bmax = red[0]; __syncthreads();
  float lsum = 0.f;
  for (int m = tid; m < N; m += 128) { float e = __expf(vals[m] - bmax); vals[m] = e; lsum += e; }
  red[tid] = lsum; __syncthreads();
  for (int s = 64; s > 0; s >>= 1) { if (tid < s) red[tid] += red[tid + s]; __syncthreads(); }
  float inv = 1.f / red[0];
  float alpha = alpha_p[0], beta = beta_p[0];
  for (int m = tid; m < N; m += 128) {
    float lp = lap[n * N + m];
    bm[n * N + m] = alpha * vals[m] * inv + beta * lp + (lp != 0.f ? 0.f : -1e9f);
  }
}

// ---------- LayerNorm (wave32 per row of 128) -> bf16 ----------
__global__ __launch_bounds__(256) void ln_kernel(
    const float* __restrict__ x, const float* __restrict__ g,
    const float* __restrict__ b, bf16_t* __restrict__ y, int rows) {
  int wid = (int)((blockIdx.x * blockDim.x + threadIdx.x) >> 5);
  int lane = threadIdx.x & 31;
  if (wid >= rows) return;
  float4v v = ((const float4v*)(x + (size_t)wid * DMODEL))[lane];
  float s = v.x + v.y + v.z + v.w;
  #pragma unroll
  for (int m = 1; m < 32; m <<= 1) s += __shfl_xor(s, m, 32);
  float mean = s * (1.f / 128.f);
  float dx = v.x - mean, dy = v.y - mean, dz = v.z - mean, dw = v.w - mean;
  float qv = dx * dx + dy * dy + dz * dz + dw * dw;
  #pragma unroll
  for (int m = 1; m < 32; m <<= 1) qv += __shfl_xor(qv, m, 32);
  float rstd = rsqrtf(qv * (1.f / 128.f) + 1e-5f);
  float4v gv = ((const float4v*)g)[lane];
  float4v bv = ((const float4v*)b)[lane];
  ushort4v o;
  o.x = f2bfbits(dx * rstd * gv.x + bv.x);
  o.y = f2bfbits(dy * rstd * gv.y + bv.y);
  o.z = f2bfbits(dz * rstd * gv.z + bv.z);
  o.w = f2bfbits(dw * rstd * gv.w + bv.w);
  ((ushort4v*)(y + (size_t)wid * DMODEL))[lane] = o;
}

// ---------- generic bf16 WMMA GEMM: C[M,N] = A[M,K] @ B[K,N] + bias ----------
// 16x64 tile per wave: A fragment reused across 4 WMMAs per K-step.
// EPI: 0 = store bf16, 1 = relu + store bf16, 2 = + resid, store fp32
template <int EPI>
__global__ __launch_bounds__(256) void gemm_kernel(
    const bf16_t* __restrict__ A, const unsigned* __restrict__ Bp,
    const float* __restrict__ bias, const float* __restrict__ resid,
    void* __restrict__ outp, int M, int N, int K) {
  int tid = threadIdx.x;
  int wid = blockIdx.x * 8 + (tid >> 5);
  int tn4 = N >> 6;                       // 64-col groups
  int ntile = (M >> 4) * tn4;
  if (wid >= ntile) return;
  int tm = wid / tn4, tc = wid % tn4;
  int lane = tid & 31, h = lane >> 4, lm = lane & 15;
  size_t row = (size_t)tm * 16 + lm;
  v8f acc[4];
  #pragma unroll
  for (int s = 0; s < 4; ++s) acc[s] = (v8f){0.f, 0.f, 0.f, 0.f, 0.f, 0.f, 0.f, 0.f};
  const bf16_t* arow = A + row * (size_t)K;
  int ksn = K >> 5;
  for (int ks = 0; ks < ksn; ++ks) {
    Frag16 a;
    a.q[0] = *(const uint4v*)(arow + ks * 32 + 16 * h);
    a.q[1] = *(const uint4v*)(arow + ks * 32 + 16 * h + 8);
    if (ks + 1 < ksn) __builtin_prefetch(arow + (ks + 1) * 32 + 8 * h, 0, 1);
    #pragma unroll
    for (int s = 0; s < 4; ++s) {
      int col = tc * 64 + s * 16 + lm;
      Frag16 b;
      const uint4v* bp = (const uint4v*)(Bp + ((size_t)(ks * N + col) * 2 + h) * 8);
      b.q[0] = bp[0]; b.q[1] = bp[1];
      acc[s] = wmma_bf16(a.v, b.v, acc[s]);
    }
  }
  #pragma unroll
  for (int s = 0; s < 4; ++s) {
    int col = tc * 64 + s * 16 + lm;
    float bi = bias[col];
    if (EPI == 2) {
      float* out = (float*)outp;
      #pragma unroll
      for (int r = 0; r < 8; ++r) {
        size_t rr = (size_t)tm * 16 + r + 8 * h;
        size_t idx = rr * (size_t)N + col;
        out[idx] = acc[s][r] + bi + resid[idx];
      }
    } else {
      bf16_t* out = (bf16_t*)outp;
      #pragma unroll
      for (int r = 0; r < 8; ++r) {
        size_t rr = (size_t)tm * 16 + r + 8 * h;
        float val = acc[s][r] + bi;
        if (EPI == 1) val = fmaxf(val, 0.f);
        out[rr * (size_t)N + col] = f2bf(val);
      }
    }
  }
}

// ---------- fused attention: per (b*t, head) block ----------
__global__ __launch_bounds__(128) void attn_kernel(
    const bf16_t* __restrict__ q, const bf16_t* __restrict__ k,
    const bf16_t* __restrict__ v, const float* __restrict__ bm,
    bf16_t* __restrict__ o) {
  const int N = NNODE;
  int bt = blockIdx.x >> 3, head = blockIdx.x & 7;
  size_t base = (size_t)bt * N * DMODEL + head * 16;
  __shared__ bf16_t kls[NPADK * 16];     // k rows [m][d]
  __shared__ bf16_t vtl[16 * NPADK];     // v transposed [d][m]
  __shared__ float  sc[4][16 * NQ];      // per-wave scores
  __shared__ bf16_t pbuf[4][16 * NQ];    // per-wave unnormalized probs (bf16)
  __shared__ float  invbuf[4][16];       // per-wave 1/rowsum
  __shared__ uint4v zpadv[2];            // 32B of zeros: pad reads, no exec masking
  int tid = threadIdx.x, w = tid >> 5, lane = tid & 31;
  int h = lane >> 4, lm = lane & 15;
  uint4v z4 = {0u, 0u, 0u, 0u};
  if (tid < 2) zpadv[tid] = z4;
  const bf16_t* zpad = (const bf16_t*)zpadv;

  // ---- K rows into LDS (async on CDNA5; sync fallback) ----
#if ATTN_ASYNC
  for (int m = tid; m < N; m += 128) {
    const bf16_t* src = k + base + (size_t)m * DMODEL;
    __builtin_amdgcn_global_load_async_to_lds_b128(
        AS1_I4(src), AS3_I4(kls + m * 16), 0, 0);
    __builtin_amdgcn_global_load_async_to_lds_b128(
        AS1_I4(src + 8), AS3_I4(kls + m * 16 + 8), 0, 0);
  }
#else
  for (int m = tid; m < N; m += 128) {
    const uint4v* ksrc = (const uint4v*)(k + base + (size_t)m * DMODEL);
    ((uint4v*)(kls + m * 16))[0] = ksrc[0];
    ((uint4v*)(kls + m * 16))[1] = ksrc[1];
  }
#endif
  // ---- V transposed into LDS + zero padding ----
  for (int m = tid; m < NPADK; m += 128) {
    if (m < N) {
      const bf16_t* vr = v + base + (size_t)m * DMODEL;
      #pragma unroll
      for (int d = 0; d < 16; ++d) vtl[d * NPADK + m] = vr[d];
    } else {
      bf16_t zb = f2bf(0.f);
      #pragma unroll
      for (int d = 0; d < 16; ++d) vtl[d * NPADK + m] = zb;
      ((uint4v*)(kls + m * 16))[0] = z4;
      ((uint4v*)(kls + m * 16))[1] = z4;
    }
  }
#if ATTN_ASYNC
  __builtin_amdgcn_s_wait_asynccnt(0);
#endif
  __syncthreads();

  float*  srow = sc[w];
  bf16_t* prow = pbuf[w];
  float*  winv = invbuf[w];
  // Score B-fragment source: h==0 lanes walk k rows; h==1 lanes (K pad 16..31)
  // always read the shared zero block -> unconditional ds loads, no exec dance.
  const bf16_t* kvbase = (h == 0) ? (kls + lm * 16) : zpad;
  int kvstep = (h == 0) ? 256 : 0;       // 16 rows * 16 elems per ct
  for (int rt = w; rt < 21; rt += 4) {
    int row = rt * 16 + lm;
    int rowc = row < N ? row : N - 1;    // clamp: OOB rows computed but never stored
    Frag16 a;
    const bf16_t* qsrc = q + base + (size_t)rowc * DMODEL;
    a.q[0] = (h == 0) ? *(const uint4v*)qsrc : z4;
    a.q[1] = (h == 0) ? *(const uint4v*)(qsrc + 8) : z4;  // dh=16 zero-padded to K=32
    // --- raw scores: (q . k^T) * scale ---
    for (int ct = 0; ct < 21; ++ct) {
      const uint4v* kp = (const uint4v*)(kvbase + ct * kvstep);
      Frag16 b;
      b.q[0] = kp[0]; b.q[1] = kp[1];
      v8f c = {0.f, 0.f, 0.f, 0.f, 0.f, 0.f, 0.f, 0.f};
      c = wmma_bf16(a.v, b.v, c);
      int colg = ct * 16 + lm;
      #pragma unroll
      for (int r = 0; r < 8; ++r)
        srow[(r + 8 * h) * NQ + colg] = c[r] * 0.25f;
    }
    // --- bias + mask + softmax per row (coalesced bias reads) ---
    for (int i = 0; i < 16; ++i) {
      int rn = rt * 16 + i;
      float mx = -3.0e38f;
      if (rn < N) {
        const float* brow = bm + (size_t)rn * N;
        for (int cix = lane; cix < NQ; cix += 32) {
          float t = (cix < N) ? (srow[i * NQ + cix] + brow[cix]) : -1e9f;
          srow[i * NQ + cix] = t;
          mx = fmaxf(mx, t);
        }
      } else {
        for (int cix = lane; cix < NQ; cix += 32) srow[i * NQ + cix] = -1e9f;
        mx = -1e9f;
      }
      #pragma unroll
      for (int mm = 1; mm < 32; mm <<= 1) mx = fmaxf(mx, __shfl_xor(mx, mm, 32));
      float sum = 0.f;
      for (int cix = lane; cix < NQ; cix += 32) {
        float e = __expf(srow[i * NQ + cix] - mx);
        sum += e;
        prow[i * NQ + cix] = f2bf(e);   // unnormalized
      }
      #pragma unroll
      for (int mm = 1; mm < 32; mm <<= 1) sum += __shfl_xor(sum, mm, 32);
      if (lane == 0) winv[i] = 1.f / sum;
    }
    // --- out = (P @ V) * rowinv ---
    v8f oc = {0.f, 0.f, 0.f, 0.f, 0.f, 0.f, 0.f, 0.f};
    for (int u = 0; u < 11; ++u) {
      Frag16 pa, vb;
      int c0 = u * 32 + 8 * h;
      pa.q[0] = *(const uint4v*)(prow + lm * NQ + c0);
      int c1 = u * 32 + 16 + 8 * h;
      const bf16_t* psrc1 = (c1 < NQ) ? (prow + lm * NQ + c1) : zpad;
      pa.q[1] = *(const uint4v*)psrc1;
      int r0 = u * 32 + 8 * h;
      vb.q[0] = *(const uint4v*)(vtl + lm * NPADK + r0);
      vb.q[1] = *(const uint4v*)(vtl + lm * NPADK + r0 + 16);
      oc = wmma_bf16(pa.v, vb.v, oc);
    }
    #pragma unroll
    for (int r = 0; r < 8; ++r) {
      int rn = rt * 16 + r + 8 * h;
      if (rn < N) o[base + (size_t)rn * DMODEL + lm] = f2bf(oc[r] * winv[r + 8 * h]);
    }
  }
}

// ---------- workspace layout (bytes) ----------
#define OFF_PWQ 0u
#define OFF_PWK 32768u
#define OFF_PWV 65536u
#define OFF_PWO 98304u
#define OFF_PW1 131072u
#define OFF_PW2 262144u
#define OFF_BM  393216u
#define OFF_XN  815872u
#define OFF_Q   8803072u
#define OFF_K   16790272u
#define OFF_V   24777472u
#define OFF_AO  32764672u
#define OFF_X1  40751872u
// h reuses XN; ffn1 reuses Q..AO (exactly 31,948,800 bytes)

extern "C" void kernel_launch(void* const* d_in, const int* in_sizes, int n_in,
                              void* d_out, int out_size, void* d_ws, size_t ws_size,
                              hipStream_t stream) {
  const float* x   = (const float*)d_in[0];
  const float* lap = (const float*)d_in[1];
  const float* emb = (const float*)d_in[2];
  const float* Wq  = (const float*)d_in[3];
  const float* bq  = (const float*)d_in[4];
  const float* Wk  = (const float*)d_in[5];
  const float* bk  = (const float*)d_in[6];
  const float* Wv  = (const float*)d_in[7];
  const float* bv  = (const float*)d_in[8];
  const float* g1  = (const float*)d_in[9];
  const float* b1  = (const float*)d_in[10];
  const float* Wo  = (const float*)d_in[11];
  const float* bo  = (const float*)d_in[12];
  const float* g2  = (const float*)d_in[13];
  const float* b2  = (const float*)d_in[14];
  const float* W1  = (const float*)d_in[15];
  const float* bf1 = (const float*)d_in[16];
  const float* W2  = (const float*)d_in[17];
  const float* bf2 = (const float*)d_in[18];
  const float* alpha = (const float*)d_in[19];
  const float* beta  = (const float*)d_in[20];

  char* ws = (char*)d_ws;
  unsigned* PWq = (unsigned*)(ws + OFF_PWQ);
  unsigned* PWk = (unsigned*)(ws + OFF_PWK);
  unsigned* PWv = (unsigned*)(ws + OFF_PWV);
  unsigned* PWo = (unsigned*)(ws + OFF_PWO);
  unsigned* PW1 = (unsigned*)(ws + OFF_PW1);
  unsigned* PW2 = (unsigned*)(ws + OFF_PW2);
  float*  BM  = (float*)(ws + OFF_BM);
  bf16_t* xn  = (bf16_t*)(ws + OFF_XN);
  bf16_t* qb  = (bf16_t*)(ws + OFF_Q);
  bf16_t* kb  = (bf16_t*)(ws + OFF_K);
  bf16_t* vb2 = (bf16_t*)(ws + OFF_V);
  bf16_t* ao  = (bf16_t*)(ws + OFF_AO);
  float*  x1  = (float*)(ws + OFF_X1);
  bf16_t* hb  = (bf16_t*)(ws + OFF_XN);  // reuse
  bf16_t* f1  = (bf16_t*)(ws + OFF_Q);   // reuse

  // 1) pack weights into WMMA B-fragment layout (bf16)
  pack_kernel<<<(4 * 128 * 16) / 256, 256, 0, stream>>>(Wq, PWq, 128, 128);
  pack_kernel<<<(4 * 128 * 16) / 256, 256, 0, stream>>>(Wk, PWk, 128, 128);
  pack_kernel<<<(4 * 128 * 16) / 256, 256, 0, stream>>>(Wv, PWv, 128, 128);
  pack_kernel<<<(4 * 128 * 16) / 256, 256, 0, stream>>>(Wo, PWo, 128, 128);
  pack_kernel<<<(4 * 512 * 16) / 256, 256, 0, stream>>>(W1, PW1, 128, 512);
  pack_kernel<<<(16 * 128 * 16) / 256, 256, 0, stream>>>(W2, PW2, 512, 128);

  // 2) combined attention bias matrix
  bias_kernel<<<NNODE, 128, 0, stream>>>(lap, emb, alpha, beta, BM);

  // 3) LN(x) -> xn (bf16)
  ln_kernel<<<NROWS / 8, 256, 0, stream>>>(x, g1, b1, xn, NROWS);

  // 4) q,k,v projections (16x64 per wave)
  int blk128 = ((NROWS / 16) * (128 / 64) + 7) / 8;  // 488
  gemm_kernel<0><<<blk128, 256, 0, stream>>>(xn, PWq, bq, nullptr, qb, NROWS, 128, 128);
  gemm_kernel<0><<<blk128, 256, 0, stream>>>(xn, PWk, bk, nullptr, kb, NROWS, 128, 128);
  gemm_kernel<0><<<blk128, 256, 0, stream>>>(xn, PWv, bv, nullptr, vb2, NROWS, 128, 128);

  // 5) fused attention (96 bt positions x 8 heads)
  attn_kernel<<<96 * 8, 128, 0, stream>>>(qb, kb, vb2, BM, ao);

  // 6) x1 = x + attn_out @ Wo + bo  (fp32)
  gemm_kernel<2><<<blk128, 256, 0, stream>>>(ao, PWo, bo, x, x1, NROWS, 128, 128);

  // 7) h = LN(x1)
  ln_kernel<<<NROWS / 8, 256, 0, stream>>>(x1, g2, b2, hb, NROWS);

  // 8) ffn1 = relu(h @ W1 + bf1)
  int blk512 = ((NROWS / 16) * (512 / 64) + 7) / 8;  // 1950
  gemm_kernel<1><<<blk512, 256, 0, stream>>>(hb, PW1, bf1, nullptr, f1, NROWS, 512, 128);

  // 9) out = x1 + ffn1 @ W2 + bf2  (fp32, final)
  gemm_kernel<2><<<blk128, 256, 0, stream>>>(f1, PW2, bf2, x1, (float*)d_out, NROWS, 128, 512);
}
